// GNN_5205500363101
// MI455X (gfx1250) — compile-verified, matrix operand order checked by reading, not run on hardware
//
#include <hip/hip_runtime.h>

#define NNODES 50000
#define NEDGES 500000
#define EPSV   1e-5f
#define DT_F   0.005f

typedef __attribute__((ext_vector_type(16))) __bf16 v16bf;
typedef __attribute__((ext_vector_type(8)))  __bf16 v8bf;
typedef __attribute__((ext_vector_type(4)))  __bf16 v4bf;
typedef __attribute__((ext_vector_type(2)))  __bf16 v2bf;
typedef __attribute__((ext_vector_type(8)))  float  v8f;

// ---------------------------------------------------------------------------
// CDNA5 WMMA fragment layout (ISA 7.12.2), V_WMMA_F32_16X16X32_BF16, wave32:
//   A (16x32): lane L (m=L&15, hi=L>>4), elem p -> A[m][8*hi+p]      (p<8)
//                                                  A[m][16+8*hi+p-8] (p>=8)
//   => a fragment is TWO contiguous 16-byte runs of a row-major tile.
//   B (32x16): same k-map with lane = output column slot.
//   C/D (16x16 f32): lane L vgpr r -> row r+8*hi, col-slot L&15.
//
// Column interleave trick: a wave's two 16-wide tiles carry the EVEN (t=0)
// and ODD (t=1) columns of its 32-column slab, i.e. lane m computes the
// adjacent column pair (cb+2m, cb+2m+1). The D->LDS store then packs both
// accumulators into one dword (paired v_cvt_pk_bf16_f32 + ds_store_b32).
// The permutation lives entirely in the weight shuffle + store epilogue;
// the activation buffer stays natural row-major for the A-fragment reads.
//
// Weights are pre-shuffled fragment-major: each lane's B fragment is 32
// contiguous bytes at Wf[((chunk*8 + wave*2 + t)*32 + lane)*16].
// ---------------------------------------------------------------------------
template <int KP, int LDA>
__device__ __forceinline__ void wmma_tile(const __bf16* As,
                                          const __bf16* __restrict__ Wf,
                                          int lane, int wave, v8f acc[2]) {
  const int m = lane & 15, hi = lane >> 4;
  v8f z = {0.f, 0.f, 0.f, 0.f, 0.f, 0.f, 0.f, 0.f};
  acc[0] = z; acc[1] = z;
#pragma unroll
  for (int ch = 0; ch < KP / 32; ++ch) {
    union { v16bf v; v8bf h[2]; } a;
    a.h[0] = *(const v8bf*)(As + m * LDA + ch * 32 + (hi << 3));        // k: 8hi..8hi+7
    a.h[1] = *(const v8bf*)(As + m * LDA + ch * 32 + 16 + (hi << 3));   // k: 16+8hi..
#pragma unroll
    for (int t = 0; t < 2; ++t) {
      const v16bf b =
          *(const v16bf*)(Wf + (((ch * 8 + wave * 2 + t) * 32 + lane) << 4));
      acc[t] = __builtin_amdgcn_wmma_f32_16x16x32_bf16(
          false, a.v, false, b, (short)0, acc[t], false, false);
    }
  }
}

// D tile -> LDS bf16 activation buffer (stride 136), bias + optional ReLU.
// Lane m holds cols (cb+2m, cb+2m+1): pack both accs into one dword store.
__device__ __forceinline__ void store_bf16(__bf16* dst, const v8f acc[2],
                                           const float* bias, int lane, int cb,
                                           bool relu) {
  const int m = lane & 15, hi = lane >> 4;
  const float2 bv = *(const float2*)(bias + cb + 2 * m);
  __bf16* colp = dst + (hi << 3) * 136 + cb + 2 * m;
#pragma unroll
  for (int r = 0; r < 8; ++r) {
    float v0 = acc[0][r] + bv.x;
    float v1 = acc[1][r] + bv.y;
    if (relu) { v0 = fmaxf(v0, 0.0f); v1 = fmaxf(v1, 0.0f); }
    v2bf pk = {(__bf16)v0, (__bf16)v1};
    *(v2bf*)(colp + r * 136) = pk;          // one ds_store_b32
  }
}

// D tile -> LDS f32 buffer (stride 132), bias + optional ReLU (float2 stores)
__device__ __forceinline__ void store_f32(float* dst, const v8f acc[2],
                                          const float* bias, int lane, int cb,
                                          bool relu) {
  const int m = lane & 15, hi = lane >> 4;
  const float2 bv = *(const float2*)(bias + cb + 2 * m);
  float* colp = dst + (hi << 3) * 132 + cb + 2 * m;
#pragma unroll
  for (int r = 0; r < 8; ++r) {
    float2 o;
    o.x = acc[0][r] + bv.x;
    o.y = acc[1][r] + bv.y;
    if (relu) { o.x = fmaxf(o.x, 0.0f); o.y = fmaxf(o.y, 0.0f); }
    *(float2*)(colp + r * 132) = o;         // one ds_store_b64
  }
}

// LayerNorm stats over 128 features, 8 threads/row, biased variance.
// Os stride 132 f32. Leaves mu_s/rs_s valid for all threads (syncs inside).
__device__ __forceinline__ void ln_rowstats(const float* Os, float* red1,
                                            float* red2, float* mu_s,
                                            float* rs_s, int tid) {
  const int r = tid >> 3, part = tid & 7;
  float s = 0.f, s2 = 0.f;
#pragma unroll
  for (int q = 0; q < 4; ++q) {
    const float4 v = *(const float4*)&Os[r * 132 + part * 16 + q * 4];
    s += v.x + v.y + v.z + v.w;
    s2 += v.x * v.x + v.y * v.y + v.z * v.z + v.w * v.w;
  }
  red1[tid] = s; red2[tid] = s2;
  __syncthreads();
  if (part < 4) { red1[tid] += red1[tid + 4]; red2[tid] += red2[tid + 4]; }
  __syncthreads();
  if (part < 2) { red1[tid] += red1[tid + 2]; red2[tid] += red2[tid + 2]; }
  __syncthreads();
  if (part == 0) {
    float mu = (red1[tid] + red1[tid + 1]) * (1.0f / 128.0f);
    float var = (red2[tid] + red2[tid + 1]) * (1.0f / 128.0f) - mu * mu;
    mu_s[r] = mu;
    rs_s[r] = rsqrtf(var + EPSV);
  }
  __syncthreads();
}

// ---------------------------------------------------------------------------
// fp32 [K x 128] -> bf16 fragment-major weights, K zero-padded to Kp.
// Column interleave: fragment slot (wave, t, n) carries column
// 32*wave + 2*n + t, matching the packed D-store above.
// ---------------------------------------------------------------------------
__global__ void shuffle_w_kernel(const float* __restrict__ src,
                                 __bf16* __restrict__ dst, int K, int Kp) {
  int f = blockIdx.x * 256 + threadIdx.x;
  if (f >= (Kp >> 5) * 256) return;
  const int lane = f & 31, ct = (f >> 5) & 7, ch = f >> 8;
  const int hi = lane >> 4, n = lane & 15;
  const int col = (ct >> 1) * 32 + 2 * n + (ct & 1);   // even/odd interleave
  v16bf out;
#pragma unroll
  for (int p = 0; p < 16; ++p) {
    int k = (ch << 5) + (hi << 3) + ((p < 8) ? p : (8 + p));  // +16 for p>=8
    out[p] = (k < K) ? (__bf16)src[(size_t)k * 128 + col] : (__bf16)0.0f;
  }
  *(v16bf*)(dst + ((size_t)f << 4)) = out;
}

// ---------------------------------------------------------------------------
// BatchNorm stats: acc[0..7]=sum, acc[8..15]=sum of squares (per feature)
// ---------------------------------------------------------------------------
__global__ __launch_bounds__(256) void bn_stats_kernel(
    const float* __restrict__ ea, float* __restrict__ acc) {
  __shared__ float s1[256], s2[256];
  const int tid = threadIdx.x;
  const int d = tid & 7;
  float a = 0.f, b = 0.f;
  for (int r = blockIdx.x * 32 + (tid >> 3); r < NEDGES; r += gridDim.x * 32) {
    float v = ea[(size_t)r * 8 + d];
    a += v; b += v * v;
  }
  s1[tid] = a; s2[tid] = b;
  __syncthreads();
  for (int off = 128; off >= 8; off >>= 1) {
    if (tid < off) { s1[tid] += s1[tid + off]; s2[tid] += s2[tid + off]; }
    __syncthreads();
  }
  if (tid < 8) {
    atomicAdd(&acc[tid], s1[tid]);
    atomicAdd(&acc[8 + tid], s2[tid]);
  }
}

// ---------------------------------------------------------------------------
// Node encoder: h = LN(MLP_16->128->128->128(x))
// ---------------------------------------------------------------------------
__global__ __launch_bounds__(128) void enc_node_kernel(
    const float* __restrict__ x, float* __restrict__ h,
    const __bf16* W0, const float* b0, const __bf16* W1, const float* b1,
    const __bf16* W2, const float* b2, const float* lng, const float* lnb) {
  __shared__ __align__(32) __bf16 As[16 * 40];
  __shared__ __align__(32) __bf16 Bs[16 * 136];
  __shared__ __align__(32) __bf16 Cs[16 * 136];
  __shared__ __align__(16) float  Os[16 * 132];
  __shared__ float red1[128], red2[128], mu_s[16], rs_s[16];
  const int tid = threadIdx.x, lane = tid & 31, wave = tid >> 5;
  const int base = blockIdx.x * 16;
  __builtin_prefetch(W1, 0, 2);
  __builtin_prefetch(W2, 0, 2);

  {  // stage [16 x 32] (K=16 zero-padded), one float4 per thread
    int r = tid >> 3, c4 = (tid & 7) * 4;
    v4bf b4 = {(__bf16)0.f, (__bf16)0.f, (__bf16)0.f, (__bf16)0.f};
    if (c4 < 16) {
      float4 v = *(const float4*)(x + (size_t)(base + r) * 16 + c4);
      b4 = (v4bf){(__bf16)v.x, (__bf16)v.y, (__bf16)v.z, (__bf16)v.w};
    }
    *(v4bf*)(As + r * 40 + c4) = b4;
  }
  __syncthreads();

  v8f acc[2];
  const int cb = wave * 32;
  wmma_tile<32, 40>(As, W0, lane, wave, acc);
  store_bf16(Bs, acc, b0, lane, cb, true);
  __syncthreads();
  wmma_tile<128, 136>(Bs, W1, lane, wave, acc);
  store_bf16(Cs, acc, b1, lane, cb, true);
  __syncthreads();
  wmma_tile<128, 136>(Cs, W2, lane, wave, acc);
  store_f32(Os, acc, b2, lane, cb, false);
  __syncthreads();
  ln_rowstats(Os, red1, red2, mu_s, rs_s, tid);

#pragma unroll
  for (int q = 0; q < 4; ++q) {
    int idx = tid + q * 128;
    int r = idx >> 5, c4 = (idx & 31) * 4;
    float4 o = *(const float4*)&Os[r * 132 + c4];
    float4 g = *(const float4*)&lng[c4];
    float4 bb = *(const float4*)&lnb[c4];
    float4 out;
    out.x = (o.x - mu_s[r]) * rs_s[r] * g.x + bb.x;
    out.y = (o.y - mu_s[r]) * rs_s[r] * g.y + bb.y;
    out.z = (o.z - mu_s[r]) * rs_s[r] * g.z + bb.z;
    out.w = (o.w - mu_s[r]) * rs_s[r] * g.w + bb.w;
    *(float4*)(h + (size_t)(base + r) * 128 + c4) = out;
  }
}

// ---------------------------------------------------------------------------
// Edge encoder: e = LN(MLP_8->128->128->128(BN(edge_attr)))
// ---------------------------------------------------------------------------
__global__ __launch_bounds__(128) void enc_edge_kernel(
    const float* __restrict__ ea, float* __restrict__ e,
    const float* __restrict__ bnacc, const float* bng, const float* bnb,
    const __bf16* W0, const float* b0, const __bf16* W1, const float* b1,
    const __bf16* W2, const float* b2, const float* lng, const float* lnb) {
  __shared__ __align__(32) __bf16 As[16 * 40];
  __shared__ __align__(32) __bf16 Bs[16 * 136];
  __shared__ __align__(32) __bf16 Cs[16 * 136];
  __shared__ __align__(16) float  Os[16 * 132];
  __shared__ float red1[128], red2[128], mu_s[16], rs_s[16];
  const int tid = threadIdx.x, lane = tid & 31, wave = tid >> 5;
  const int base = blockIdx.x * 16;
  __builtin_prefetch(W1, 0, 2);
  __builtin_prefetch(W2, 0, 2);

  {  // stage BN(edge_attr) into [16 x 32] (K=8 zero-padded)
    int r = tid >> 3, c4 = (tid & 7) * 4;
    v4bf b4 = {(__bf16)0.f, (__bf16)0.f, (__bf16)0.f, (__bf16)0.f};
    if (c4 < 8) {
      float4 v = *(const float4*)(ea + (size_t)(base + r) * 8 + c4);
      float vv[4] = {v.x, v.y, v.z, v.w};
#pragma unroll
      for (int j = 0; j < 4; ++j) {
        int c = c4 + j;
        float mu = bnacc[c] * (1.0f / NEDGES);
        float var = bnacc[8 + c] * (1.0f / NEDGES) - mu * mu;
        b4[j] = (__bf16)((vv[j] - mu) * rsqrtf(var + EPSV) * bng[c] + bnb[c]);
      }
    }
    *(v4bf*)(As + r * 40 + c4) = b4;
  }
  __syncthreads();

  v8f acc[2];
  const int cb = wave * 32;
  wmma_tile<32, 40>(As, W0, lane, wave, acc);
  store_bf16(Bs, acc, b0, lane, cb, true);
  __syncthreads();
  wmma_tile<128, 136>(Bs, W1, lane, wave, acc);
  store_bf16(Cs, acc, b1, lane, cb, true);
  __syncthreads();
  wmma_tile<128, 136>(Cs, W2, lane, wave, acc);
  store_f32(Os, acc, b2, lane, cb, false);
  __syncthreads();
  ln_rowstats(Os, red1, red2, mu_s, rs_s, tid);

#pragma unroll
  for (int q = 0; q < 4; ++q) {
    int idx = tid + q * 128;
    int r = idx >> 5, c4 = (idx & 31) * 4;
    float4 o = *(const float4*)&Os[r * 132 + c4];
    float4 g = *(const float4*)&lng[c4];
    float4 bb = *(const float4*)&lnb[c4];
    float4 out;
    out.x = (o.x - mu_s[r]) * rs_s[r] * g.x + bb.x;
    out.y = (o.y - mu_s[r]) * rs_s[r] * g.y + bb.y;
    out.z = (o.z - mu_s[r]) * rs_s[r] * g.z + bb.z;
    out.w = (o.w - mu_s[r]) * rs_s[r] * g.w + bb.w;
    *(float4*)(e + (size_t)(base + r) * 128 + c4) = out;
  }
}

// ---------------------------------------------------------------------------
// Edge processor: e += LN(MLP_384->128([h[row],h[col],e])); fused scatter
// ---------------------------------------------------------------------------
__global__ __launch_bounds__(128) void edge_kernel(
    const float* __restrict__ h, float* __restrict__ e, float* __restrict__ agg,
    const int* __restrict__ row, const int* __restrict__ col,
    const __bf16* W0, const float* b0, const __bf16* W1, const float* b1,
    const __bf16* W2, const float* b2, const float* lng, const float* lnb) {
  __shared__ __align__(32) __bf16 As[16 * 392];
  __shared__ __align__(32) __bf16 Bs[16 * 136];
  __shared__ __align__(32) __bf16 Cs[16 * 136];
  __shared__ __align__(16) float  Os[16 * 132];
  __shared__ float red1[128], red2[128], mu_s[16], rs_s[16];
  __shared__ int rows_s[16], cols_s[16];
  const int tid = threadIdx.x, lane = tid & 31, wave = tid >> 5;
  const int base = blockIdx.x * 16;
  __builtin_prefetch(W0, 0, 2);
  __builtin_prefetch(W1, 0, 2);
  __builtin_prefetch(W2, 0, 2);

  if (tid < 16) { rows_s[tid] = row[base + tid]; cols_s[tid] = col[base + tid]; }
  __syncthreads();

  // stage [16 x 384] concat tile; one float4 -> 4 bf16 per step
#pragma unroll
  for (int q = 0; q < 12; ++q) {
    int idx = tid + q * 128;           // 16 * 96 float4 slots
    int r = idx / 96, c4 = (idx - r * 96) * 4;
    const float* src;
    if (c4 < 128)      src = h + (size_t)rows_s[r] * 128 + c4;
    else if (c4 < 256) src = h + (size_t)cols_s[r] * 128 + (c4 - 128);
    else               src = e + (size_t)(base + r) * 128 + (c4 - 256);
    float4 v = *(const float4*)src;
    v4bf b4 = {(__bf16)v.x, (__bf16)v.y, (__bf16)v.z, (__bf16)v.w};
    *(v4bf*)(As + r * 392 + c4) = b4;
  }
  __syncthreads();

  v8f acc[2];
  const int cb = wave * 32;
  wmma_tile<384, 392>(As, W0, lane, wave, acc);
  store_bf16(Bs, acc, b0, lane, cb, true);
  __syncthreads();
  wmma_tile<128, 136>(Bs, W1, lane, wave, acc);
  store_bf16(Cs, acc, b1, lane, cb, true);
  __syncthreads();
  wmma_tile<128, 136>(Cs, W2, lane, wave, acc);
  store_f32(Os, acc, b2, lane, cb, false);
  __syncthreads();
  ln_rowstats(Os, red1, red2, mu_s, rs_s, tid);

#pragma unroll
  for (int q = 0; q < 4; ++q) {
    int idx = tid + q * 128;
    int r = idx >> 5, c4 = (idx & 31) * 4;
    int ed = base + r;
    float4 o = *(const float4*)&Os[r * 132 + c4];
    float4 g = *(const float4*)&lng[c4];
    float4 bb = *(const float4*)&lnb[c4];
    float4 res = *(const float4*)(e + (size_t)ed * 128 + c4);
    float4 out;
    out.x = (o.x - mu_s[r]) * rs_s[r] * g.x + bb.x + res.x;
    out.y = (o.y - mu_s[r]) * rs_s[r] * g.y + bb.y + res.y;
    out.z = (o.z - mu_s[r]) * rs_s[r] * g.z + bb.z + res.z;
    out.w = (o.w - mu_s[r]) * rs_s[r] * g.w + bb.w + res.w;
    *(float4*)(e + (size_t)ed * 128 + c4) = out;
    float* ag = agg + (size_t)cols_s[r] * 128 + c4;   // fused segment_sum
    atomicAdd(ag + 0, out.x);
    atomicAdd(ag + 1, out.y);
    atomicAdd(ag + 2, out.z);
    atomicAdd(ag + 3, out.w);
  }
}

// ---------------------------------------------------------------------------
// Node processor: h += LN(MLP_256->128([h, agg]))
// ---------------------------------------------------------------------------
__global__ __launch_bounds__(128) void node_kernel(
    float* __restrict__ h, const float* __restrict__ agg,
    const __bf16* W0, const float* b0, const __bf16* W1, const float* b1,
    const __bf16* W2, const float* b2, const float* lng, const float* lnb) {
  __shared__ __align__(32) __bf16 As[16 * 264];
  __shared__ __align__(32) __bf16 Bs[16 * 136];
  __shared__ __align__(32) __bf16 Cs[16 * 136];
  __shared__ __align__(16) float  Os[16 * 132];
  __shared__ float red1[128], red2[128], mu_s[16], rs_s[16];
  const int tid = threadIdx.x, lane = tid & 31, wave = tid >> 5;
  const int base = blockIdx.x * 16;
  __builtin_prefetch(W0, 0, 2);
  __builtin_prefetch(W1, 0, 2);
  __builtin_prefetch(W2, 0, 2);

#pragma unroll
  for (int q = 0; q < 8; ++q) {
    int idx = tid + q * 128;           // 16 * 64 float4 slots
    int r = idx >> 6, c4 = (idx & 63) * 4;
    const float* src = (c4 < 128) ? (h + (size_t)(base + r) * 128 + c4)
                                  : (agg + (size_t)(base + r) * 128 + (c4 - 128));
    float4 v = *(const float4*)src;
    v4bf b4 = {(__bf16)v.x, (__bf16)v.y, (__bf16)v.z, (__bf16)v.w};
    *(v4bf*)(As + r * 264 + c4) = b4;
  }
  __syncthreads();

  v8f acc[2];
  const int cb = wave * 32;
  wmma_tile<256, 264>(As, W0, lane, wave, acc);
  store_bf16(Bs, acc, b0, lane, cb, true);
  __syncthreads();
  wmma_tile<128, 136>(Bs, W1, lane, wave, acc);
  store_bf16(Cs, acc, b1, lane, cb, true);
  __syncthreads();
  wmma_tile<128, 136>(Cs, W2, lane, wave, acc);
  store_f32(Os, acc, b2, lane, cb, false);
  __syncthreads();
  ln_rowstats(Os, red1, red2, mu_s, rs_s, tid);

#pragma unroll
  for (int q = 0; q < 4; ++q) {
    int idx = tid + q * 128;
    int r = idx >> 5, c4 = (idx & 31) * 4;
    float4 o = *(const float4*)&Os[r * 132 + c4];
    float4 g = *(const float4*)&lng[c4];
    float4 bb = *(const float4*)&lnb[c4];
    float4 res = *(const float4*)(h + (size_t)(base + r) * 128 + c4);
    float4 out;
    out.x = (o.x - mu_s[r]) * rs_s[r] * g.x + bb.x + res.x;
    out.y = (o.y - mu_s[r]) * rs_s[r] * g.y + bb.y + res.y;
    out.z = (o.z - mu_s[r]) * rs_s[r] * g.z + bb.z + res.z;
    out.w = (o.w - mu_s[r]) * rs_s[r] * g.w + bb.w + res.w;
    *(float4*)(h + (size_t)(base + r) * 128 + c4) = out;
  }
}

// ---------------------------------------------------------------------------
// Decoder: out = MLP_128->128->128->1(h) * DT + x[:,0]
// ---------------------------------------------------------------------------
__global__ __launch_bounds__(128) void dec_kernel(
    const float* __restrict__ h, const float* __restrict__ x,
    const __bf16* W0, const float* b0, const __bf16* W1, const float* b1,
    const float* __restrict__ W2, const float* __restrict__ b2,
    float* __restrict__ out) {
  __shared__ __align__(32) __bf16 As[16 * 136];
  __shared__ __align__(32) __bf16 Bs[16 * 136];
  __shared__ __align__(16) float  Os[16 * 132];
  __shared__ float red[128];
  const int tid = threadIdx.x, lane = tid & 31, wave = tid >> 5;
  const int base = blockIdx.x * 16;
  __builtin_prefetch(W1, 0, 2);

#pragma unroll
  for (int q = 0; q < 4; ++q) {
    int idx = tid + q * 128;           // 16 * 32 float4 slots
    int r = idx >> 5, c4 = (idx & 31) * 4;
    float4 v = *(const float4*)(h + (size_t)(base + r) * 128 + c4);
    v4bf b4 = {(__bf16)v.x, (__bf16)v.y, (__bf16)v.z, (__bf16)v.w};
    *(v4bf*)(As + r * 136 + c4) = b4;
  }
  __syncthreads();

  v8f acc[2];
  const int cb = wave * 32;
  wmma_tile<128, 136>(As, W0, lane, wave, acc);
  store_bf16(Bs, acc, b0, lane, cb, true);
  __syncthreads();
  wmma_tile<128, 136>(Bs, W1, lane, wave, acc);
  store_f32(Os, acc, b1, lane, cb, true);   // ReLU after layer 1
  __syncthreads();

  // final 128 -> 1 dot product, 8 threads per row
  const int r = tid >> 3, part = tid & 7;
  float s = 0.f;
#pragma unroll
  for (int q = 0; q < 4; ++q) {
    float4 o = *(const float4*)&Os[r * 132 + part * 16 + q * 4];
    float4 w = *(const float4*)&W2[part * 16 + q * 4];
    s += o.x * w.x + o.y * w.y + o.z * w.z + o.w * w.w;
  }
  red[tid] = s;
  __syncthreads();
  if (part < 4) red[tid] += red[tid + 4];
  __syncthreads();
  if (part < 2) red[tid] += red[tid + 2];
  __syncthreads();
  if (part == 0) {
    float v = red[tid] + red[tid + 1] + b2[0];
    out[base + r] = v * DT_F + x[(size_t)(base + r) * 16 + 0];
  }
}

// ---------------------------------------------------------------------------
// Host launcher
// ---------------------------------------------------------------------------
struct MLPp { const float *W0, *b0, *W1, *b1, *W2, *b2, *g, *bt; };

extern "C" void kernel_launch(void* const* d_in, const int* in_sizes, int n_in,
                              void* d_out, int out_size, void* d_ws, size_t ws_size,
                              hipStream_t stream) {
  (void)out_size; (void)ws_size;
  // --- locate data tensors by unique sizes; collect param leaves in order ---
  const float* x = nullptr; const int* ei = nullptr; const float* ea = nullptr;
  int pidx[160]; int np = 0;
  for (int i = 0; i < n_in; ++i) {
    int s = in_sizes[i];
    if (s == NNODES * 16)         x  = (const float*)d_in[i];
    else if (s == 2 * NEDGES)     ei = (const int*)d_in[i];
    else if (s == NEDGES * 8)     ea = (const float*)d_in[i];
    else if (np < 160)            pidx[np++] = i;
  }
  if (!x || !ei || !ea) return;

  int cur = 0;
  auto nextp = [&]() -> const float* { return (const float*)d_in[pidx[cur++]]; };
  auto rdMLP = [&](bool ln) -> MLPp {
    MLPp m;
    m.W0 = nextp(); m.b0 = nextp(); m.W1 = nextp(); m.b1 = nextp();
    m.W2 = nextp(); m.b2 = nextp();
    if (ln) { m.g = nextp(); m.bt = nextp(); } else { m.g = nullptr; m.bt = nullptr; }
    return m;
  };

  const float *bng = nullptr, *bnb = nullptr;
  MLPp encN, encE, dec, blkE[5], blkN[5];
  // insertion order: bn(8) first; jax-sorted pytree order: blocks (edge W0=49152) first
  bool sortedOrder = (np > 0 && in_sizes[pidx[0]] == 49152);
  if (!sortedOrder) {
    bng = nextp(); bnb = nextp();
    encN = rdMLP(true); encE = rdMLP(true);
    for (int i = 0; i < 5; ++i) { blkE[i] = rdMLP(true); blkN[i] = rdMLP(true); }
    dec = rdMLP(false);
  } else {
    for (int i = 0; i < 5; ++i) { blkE[i] = rdMLP(true); blkN[i] = rdMLP(true); }
    bng = nextp(); bnb = nextp();
    dec = rdMLP(false);
    encE = rdMLP(true); encN = rdMLP(true);
  }

  // --- workspace carve (~310 MB; e kept fp32 for residual/scatter accuracy) ---
  char* ws = (char*)d_ws;
  float* h    = (float*)ws;  ws += (size_t)NNODES * 128 * 4;
  float* agg  = (float*)ws;  ws += (size_t)NNODES * 128 * 4;
  float* e    = (float*)ws;  ws += (size_t)NEDGES * 128 * 4;
  float* bnacc = (float*)ws; ws += 256;
  __bf16* arena = (__bf16*)ws;

  __bf16* encN_W0 = arena;              // 32x128 (K=16 padded)
  __bf16* encN_W1 = encN_W0 + 32 * 128;
  __bf16* encN_W2 = encN_W1 + 16384;
  __bf16* encE_W0 = encN_W2 + 16384;    // 32x128 (K=8 padded)
  __bf16* encE_W1 = encE_W0 + 32 * 128;
  __bf16* encE_W2 = encE_W1 + 16384;
  __bf16* p = encE_W2 + 16384;
  __bf16 *eW0[5], *eW1[5], *eW2[5], *nW0[5], *nW1[5], *nW2[5];
  for (int i = 0; i < 5; ++i) {
    eW0[i] = p; p += 49152; eW1[i] = p; p += 16384; eW2[i] = p; p += 16384;
    nW0[i] = p; p += 32768; nW1[i] = p; p += 16384; nW2[i] = p; p += 16384;
  }
  __bf16* dW0 = p; p += 16384;
  __bf16* dW1 = p; p += 16384;

  auto shuf = [&](const float* src, __bf16* dst, int K, int Kp) {
    int total = (Kp >> 5) * 256;       // fragments
    shuffle_w_kernel<<<(total + 255) / 256, 256, 0, stream>>>(src, dst, K, Kp);
  };

  // --- weight conversion fp32 -> bf16 fragment-major (padded) ---
  shuf(encN.W0, encN_W0, 16, 32);  shuf(encN.W1, encN_W1, 128, 128); shuf(encN.W2, encN_W2, 128, 128);
  shuf(encE.W0, encE_W0, 8, 32);   shuf(encE.W1, encE_W1, 128, 128); shuf(encE.W2, encE_W2, 128, 128);
  for (int i = 0; i < 5; ++i) {
    shuf(blkE[i].W0, eW0[i], 384, 384); shuf(blkE[i].W1, eW1[i], 128, 128); shuf(blkE[i].W2, eW2[i], 128, 128);
    shuf(blkN[i].W0, nW0[i], 256, 256); shuf(blkN[i].W1, nW1[i], 128, 128); shuf(blkN[i].W2, nW2[i], 128, 128);
  }
  shuf(dec.W0, dW0, 128, 128); shuf(dec.W1, dW1, 128, 128);

  // --- BatchNorm stats ---
  hipMemsetAsync(bnacc, 0, 64, stream);
  bn_stats_kernel<<<256, 256, 0, stream>>>(ea, bnacc);

  // --- encoders ---
  enc_node_kernel<<<NNODES / 16, 128, 0, stream>>>(
      x, h, encN_W0, encN.b0, encN_W1, encN.b1, encN_W2, encN.b2, encN.g, encN.bt);
  enc_edge_kernel<<<NEDGES / 16, 128, 0, stream>>>(
      ea, e, bnacc, bng, bnb,
      encE_W0, encE.b0, encE_W1, encE.b1, encE_W2, encE.b2, encE.g, encE.bt);

  // --- message passing ---
  const int* rowp = ei;
  const int* colp = ei + NEDGES;
  for (int it = 0; it < 5; ++it) {
    hipMemsetAsync(agg, 0, (size_t)NNODES * 128 * 4, stream);
    edge_kernel<<<NEDGES / 16, 128, 0, stream>>>(
        h, e, agg, rowp, colp,
        eW0[it], blkE[it].b0, eW1[it], blkE[it].b1, eW2[it], blkE[it].b2,
        blkE[it].g, blkE[it].bt);
    node_kernel<<<NNODES / 16, 128, 0, stream>>>(
        h, agg,
        nW0[it], blkN[it].b0, nW1[it], blkN[it].b1, nW2[it], blkN[it].b2,
        blkN[it].g, blkN[it].bt);
  }

  // --- decoder ---
  dec_kernel<<<NNODES / 16, 128, 0, stream>>>(
      h, x, dW0, dec.b0, dW1, dec.b1, dec.W2, dec.b2, (float*)d_out);
}
